// WindowAttBlock_6347961664129
// MI455X (gfx1250) — compile-verified, matrix operand order checked by reading, not run on hardware
//
#include <hip/hip_runtime.h>
#include <hip/hip_bf16.h>

#define DEV __device__ __forceinline__

typedef __bf16 bf16;
typedef __attribute__((ext_vector_type(16))) __bf16 bf16x16;
typedef __attribute__((ext_vector_type(8)))  __bf16 bf16x8;
typedef __attribute__((ext_vector_type(8)))  float  f32x8;
typedef __attribute__((ext_vector_type(4)))  unsigned int u32x4;
typedef __attribute__((ext_vector_type(8)))  int i32x8;
typedef __attribute__((ext_vector_type(4)))  int i32x4;

// problem dims
constexpr int B_ = 4, T_ = 8, P_ = 32, N_ = 64, D_ = 256, H_ = 8, M_ = 1024;
constexpr int HD = D_ / H_;  // 32
constexpr float EPS = 1e-5f;

DEV int lane_() { return threadIdx.x & 31; }
DEV int wave_() { return threadIdx.x >> 5; }

DEV f32x8 zero8() {
  f32x8 z = {0.f, 0.f, 0.f, 0.f, 0.f, 0.f, 0.f, 0.f};
  return z;
}

DEV f32x8 wmma_bf16(bf16x16 a, bf16x16 b, f32x8 c) {
  // (neg_a, A, neg_b, B, c_mod, C, reuse_a, reuse_b)
  return __builtin_amdgcn_wmma_f32_16x16x32_bf16(false, a, false, b, (short)0, c,
                                                 false, false);
}

DEV bf16x16 combine16(bf16x8 lo, bf16x8 hi) {
  bf16x16 r;
#pragma unroll
  for (int e = 0; e < 8; ++e) { r[e] = lo[e]; r[e + 8] = hi[e]; }
  return r;
}

// ---- fragment loaders (layouts per CDNA5 ISA 7.12.2, wave32) ----
// A 16x32 bf16 from LDS row-major: two 16B vector loads (ds_load_b128)
DEV bf16x16 afrag_lds(const bf16* base, int ld, int m0, int k0) {
  int l = lane_();
  const bf16* p = base + (m0 + (l & 15)) * ld + k0 + ((l >> 4) << 3);
  return combine16(*(const bf16x8*)p, *(const bf16x8*)(p + 16));
}
// B 32x16 bf16, element(k,n) = base[(k0+k)*ld + n0+n]  (strided; V operand only)
DEV bf16x16 bfrag_lds_kn(const bf16* base, int ld, int k0, int n0) {
  int l = lane_();
  int n = n0 + (l & 15);
  int kb = k0 + ((l >> 4) << 3);
  bf16x16 b;
#pragma unroll
  for (int e = 0; e < 8; ++e) {
    b[e]     = base[(kb + e) * ld + n];
    b[e + 8] = base[(kb + 16 + e) * ld + n];
  }
  return b;
}
// B 32x16 bf16, element(k,n) = base[(n0+n)*ld + k0+k] (transposed src, vectorized)
DEV bf16x16 bfrag_lds_nk(const bf16* base, int ld, int k0, int n0) {
  int l = lane_();
  const bf16* p = base + (n0 + (l & 15)) * ld + k0 + ((l >> 4) << 3);
  return combine16(*(const bf16x8*)p, *(const bf16x8*)(p + 16));
}
// B fragment from pre-transposed bf16 weights Wt[N][K]: 2x global_load_b128
DEV bf16x16 bfrag_wt(const bf16* Wt, int ldk, int k0, int n0) {
  int l = lane_();
  const bf16* p = Wt + (size_t)(n0 + (l & 15)) * ldk + k0 + ((l >> 4) << 3);
  return combine16(*(const bf16x8*)p, *(const bf16x8*)(p + 16));
}
// C/D fragment store to LDS as bf16: lane n = l&15, rows m0 + 8*(l>>4) + r
DEV void cstore_lds(bf16* base, int ld, int m0, int n0, f32x8 c) {
  int l = lane_();
  bf16* p = base + (m0 + ((l >> 4) << 3)) * ld + n0 + (l & 15);
#pragma unroll
  for (int r = 0; r < 8; ++r) p[r * ld] = (bf16)c[r];
}

DEV float gelu_(float x) { return 0.5f * x * (1.0f + erff(x * 0.70710678118f)); }

// ---- Tensor Data Mover: 2D f32 tile (rows x cols, packed) global -> LDS ----
#if defined(__gfx1250__) && __has_builtin(__builtin_amdgcn_tensor_load_to_lds)
#define HAVE_TDM 1
#else
#define HAVE_TDM 0
#endif

#if HAVE_TDM
DEV void tdm_load_2d_f32(void* lds_dst, const float* gsrc, int rows, int cols) {
  // D# per CDNA5 ISA ch.8: group0 = {count/flags, lds_addr, global_addr, type=2}
  unsigned lds_off = (unsigned)(unsigned long long)(uintptr_t)lds_dst;
  unsigned long long ga = (unsigned long long)(uintptr_t)gsrc;
  u32x4 g0;
  g0[0] = 1u;                                                // count=1, user
  g0[1] = lds_off;                                           // lds_addr
  g0[2] = (unsigned)(ga & 0xffffffffu);                      // global_addr lo
  g0[3] = (unsigned)((ga >> 32) & 0x01ffffffu) | (2u << 30); // addr hi, type=2
  i32x8 g1;
  g1[0] = 2 << 16;            // data_size=4B, workgroup_mask=0 (not in cluster)
  g1[1] = cols << 16;         // tensor_dim0[15:0]
  g1[2] = (rows << 16);       // tensor_dim0 hi=0, tensor_dim1[15:0]=rows
  g1[3] = cols << 16;         // tile_dim0
  g1[4] = rows;               // tile_dim1, tile_dim2=0
  g1[5] = cols;               // tensor_dim0_stride lo
  g1[6] = 0;                  // stride hi / dim1_stride
  g1[7] = 0;
  i32x4 gz = {0, 0, 0, 0};    // groups 2/3 unused (2D tensor)
#if __clang_major__ >= 23
  i32x8 gz8 = {0, 0, 0, 0, 0, 0, 0, 0};
  __builtin_amdgcn_tensor_load_to_lds(g0, g1, gz, gz, gz8, 0);
#else
  __builtin_amdgcn_tensor_load_to_lds(g0, g1, gz, gz, 0);
#endif
}
#endif

// stage a packed rows x cols f32 tile into LDS (TDM when available)
DEV void stage_tile_f32(float* xs, const float* gsrc, int rows, int cols) {
#if HAVE_TDM
  if (threadIdx.x == 0) tdm_load_2d_f32(xs, gsrc, rows, cols);
  __builtin_amdgcn_s_wait_tensorcnt((short)0);
#else
  for (int i = threadIdx.x; i < rows * cols; i += 256) xs[i] = gsrc[i];
#endif
  __syncthreads();
}

// =====================================================================
// Weight prep: W[K][N] fp32 -> Wt[N][K] bf16 (transpose + convert, once)
// =====================================================================
__global__ __launch_bounds__(256) void k_wt(const float* __restrict__ w,
                                            bf16* __restrict__ wt, int K, int Nn) {
  int idx = blockIdx.x * 256 + threadIdx.x;
  if (idx >= K * Nn) return;
  int n = idx / K, k = idx - n * K;
  wt[idx] = (bf16)w[(size_t)k * Nn + n];
}

// =====================================================================
// Kernel A: per-window LN1 + windowed attention (+adj bias) + mask*residual
// block = 256 threads (8 waves), one (b,t,p) window of 64 tokens x 256
// =====================================================================
__global__ __launch_bounds__(256) void k_win_attn(
    const float* __restrict__ x, const float* __restrict__ mask,
    const float* __restrict__ adj, const float* __restrict__ g1,
    const float* __restrict__ b1, const bf16* __restrict__ wt_qkv,
    const float* __restrict__ qkv_b, const bf16* __restrict__ wt_proj,
    const float* __restrict__ proj_b, float* __restrict__ out) {
  extern __shared__ __align__(16) char smem[];
  float* xs = (float*)smem;                     // 64x256 f32 (64KB) via TDM
  bf16* yb  = (bf16*)(smem + 65536);            // 64x256 bf16 (32KB), reused as O
  bf16* qkv = (bf16*)(smem + 98304);            // 64x768 bf16 (96KB)
  bf16* Sa  = (bf16*)(smem + 196608) + wave_() * N_ * N_;  // 8KB per wave

  int blk = blockIdx.x;
  int p = blk & (P_ - 1);
  size_t wbase = (size_t)blk * N_ * D_;
  int lane = lane_(), wave = wave_();

  if (threadIdx.x == 0) {
    __builtin_prefetch(wt_qkv, 0, 3);
    __builtin_prefetch(wt_proj, 0, 3);
  }
  stage_tile_f32(xs, x + wbase, N_, D_);

  // ---- LayerNorm(dn1) from LDS -> yb (bf16) ----
  for (int row = wave; row < N_; row += 8) {
    float v[8], s = 0.f, s2 = 0.f;
    const float* xr = xs + row * D_ + lane * 8;
#pragma unroll
    for (int j = 0; j < 8; ++j) { v[j] = xr[j]; s += v[j]; s2 += v[j] * v[j]; }
#pragma unroll
    for (int m = 16; m >= 1; m >>= 1) {
      s += __shfl_xor(s, m, 32);
      s2 += __shfl_xor(s2, m, 32);
    }
    float mu = s * (1.f / D_);
    float rs = rsqrtf(s2 * (1.f / D_) - mu * mu + EPS);
#pragma unroll
    for (int j = 0; j < 8; ++j) {
      int c = lane * 8 + j;
      yb[row * D_ + c] = (bf16)((v[j] - mu) * rs * g1[c] + b1[c]);
    }
  }
  __syncthreads();

  // ---- QKV GEMM: (64x256)@(256x768), 4x48 tiles over 8 waves ----
  for (int t = wave; t < 4 * 48; t += 8) {
    int tm = t & 3, tn = t >> 2;
    f32x8 acc = zero8();
#pragma unroll
    for (int k = 0; k < 8; ++k)
      acc = wmma_bf16(afrag_lds(yb, D_, tm * 16, k * 32),
                      bfrag_wt(wt_qkv, D_, k * 32, tn * 16), acc);
    int col = tn * 16 + (lane & 15);
    int rb = tm * 16 + ((lane >> 4) << 3);
    float qb = qkv_b[col];
#pragma unroll
    for (int r = 0; r < 8; ++r)
      qkv[(rb + r) * (3 * D_) + col] = (bf16)(acc[r] + qb);
  }
  __syncthreads();

  // ---- attention: wave == head (8 heads, head_dim 32 == one K-step) ----
  {
    int h = wave;
    f32x8 sacc[4][4];
    {
      bf16x16 afr[4];
#pragma unroll
      for (int tm = 0; tm < 4; ++tm)
        afr[tm] = afrag_lds(qkv, 3 * D_, tm * 16, h * HD);  // Q slice
#pragma unroll
      for (int tn = 0; tn < 4; ++tn) {
        bf16x16 bfr = bfrag_lds_nk(qkv, 3 * D_, D_ + h * HD, tn * 16);  // K^T
#pragma unroll
        for (int tm = 0; tm < 4; ++tm)
          sacc[tm][tn] = wmma_bf16(afr[tm], bfr, zero8());
      }
    }
    const float scale = 0.17677669529663687f;  // 1/sqrt(32)
#pragma unroll
    for (int tm = 0; tm < 4; ++tm) {
#pragma unroll
      for (int r = 0; r < 8; ++r) {
        int row = tm * 16 + r + ((lane >> 4) << 3);
        float vals[4];
#pragma unroll
        for (int tn = 0; tn < 4; ++tn) {
          int col = tn * 16 + (lane & 15);
          vals[tn] = sacc[tm][tn][r] * scale + logf(adj[row * N_ + col] + 1e-6f);
        }
        float mx = fmaxf(fmaxf(vals[0], vals[1]), fmaxf(vals[2], vals[3]));
#pragma unroll
        for (int m = 8; m >= 1; m >>= 1) mx = fmaxf(mx, __shfl_xor(mx, m, 32));
        float sum = 0.f;
#pragma unroll
        for (int tn = 0; tn < 4; ++tn) {
          vals[tn] = expf(vals[tn] - mx);
          sum += vals[tn];
        }
#pragma unroll
        for (int m = 8; m >= 1; m >>= 1) sum += __shfl_xor(sum, m, 32);
        float inv = 1.f / sum;
#pragma unroll
        for (int tn = 0; tn < 4; ++tn)
          Sa[row * N_ + tn * 16 + (lane & 15)] = (bf16)(vals[tn] * inv);
      }
    }
    // O = A(64x64) @ V(64x32); V-fragments hoisted (4 unique, reused by 4 tm)
    bf16x16 bv[2][2];
#pragma unroll
    for (int tn = 0; tn < 2; ++tn)
#pragma unroll
      for (int ks = 0; ks < 2; ++ks)
        bv[tn][ks] =
            bfrag_lds_kn(qkv, 3 * D_, ks * 32, 2 * D_ + h * HD + tn * 16);
#pragma unroll
    for (int tm = 0; tm < 4; ++tm) {
      bf16x16 as0 = afrag_lds(Sa, N_, tm * 16, 0);
      bf16x16 as1 = afrag_lds(Sa, N_, tm * 16, 32);
#pragma unroll
      for (int tn = 0; tn < 2; ++tn) {
        f32x8 acc = wmma_bf16(as0, bv[tn][0], zero8());
        acc = wmma_bf16(as1, bv[tn][1], acc);
        cstore_lds(yb, D_, tm * 16, h * HD + tn * 16, acc);
      }
    }
  }
  __syncthreads();

  // ---- proj (256x256) + mask*residual (residual from LDS xs) -> out ----
  const float* mrow = mask + p * N_;
  for (int t = wave; t < 64; t += 8) {
    int tm = t & 3, tn = t >> 2;
    f32x8 acc = zero8();
#pragma unroll
    for (int k = 0; k < 8; ++k)
      acc = wmma_bf16(afrag_lds(yb, D_, tm * 16, k * 32),
                      bfrag_wt(wt_proj, D_, k * 32, tn * 16), acc);
    int col = tn * 16 + (lane & 15);
    int rb = tm * 16 + ((lane >> 4) << 3);
    float pb = proj_b[col];
#pragma unroll
    for (int r = 0; r < 8; ++r) {
      int row = rb + r;
      out[wbase + (size_t)row * D_ + col] =
          xs[row * D_ + col] + (acc[r] + pb) * mrow[row];
    }
  }
}

// =====================================================================
// Kernel B: per-window LN2 + MLP (256->1024->256, exact GELU) + mask*residual
// =====================================================================
__global__ __launch_bounds__(256) void k_win_mlp(
    float* __restrict__ x5, const float* __restrict__ mask,
    const float* __restrict__ g2, const float* __restrict__ b2,
    const bf16* __restrict__ wt_fc1, const float* __restrict__ fc1_b,
    const bf16* __restrict__ wt_fc2, const float* __restrict__ fc2_b) {
  extern __shared__ __align__(16) char smem[];
  float* xs = (float*)smem;           // 64x256 f32 (64KB) via TDM
  bf16* yb = (bf16*)(smem + 65536);   // 64x256 (32KB)
  bf16* hc = (bf16*)(smem + 98304);   // 64x256 chunk (32KB)
  int blk = blockIdx.x;
  int p = blk & (P_ - 1);
  size_t wbase = (size_t)blk * N_ * D_;
  int lane = lane_(), wave = wave_();

  if (threadIdx.x == 0) {
    __builtin_prefetch(wt_fc1, 0, 3);
    __builtin_prefetch(wt_fc2, 0, 3);
  }
  stage_tile_f32(xs, x5 + wbase, N_, D_);

  for (int row = wave; row < N_; row += 8) {
    float v[8], s = 0.f, s2 = 0.f;
    const float* xr = xs + row * D_ + lane * 8;
#pragma unroll
    for (int j = 0; j < 8; ++j) { v[j] = xr[j]; s += v[j]; s2 += v[j] * v[j]; }
#pragma unroll
    for (int m = 16; m >= 1; m >>= 1) {
      s += __shfl_xor(s, m, 32);
      s2 += __shfl_xor(s2, m, 32);
    }
    float mu = s * (1.f / D_);
    float rs = rsqrtf(s2 * (1.f / D_) - mu * mu + EPS);
#pragma unroll
    for (int j = 0; j < 8; ++j) {
      int c = lane * 8 + j;
      yb[row * D_ + c] = (bf16)((v[j] - mu) * rs * g2[c] + b2[c]);
    }
  }
  __syncthreads();

  f32x8 oacc[8];
#pragma unroll
  for (int i = 0; i < 8; ++i) oacc[i] = zero8();

  for (int c = 0; c < 4; ++c) {  // M chunks of 256
    for (int t = wave; t < 64; t += 8) {
      int tm = t & 3, tn = t >> 2;
      f32x8 acc = zero8();
#pragma unroll
      for (int k = 0; k < 8; ++k)
        acc = wmma_bf16(afrag_lds(yb, D_, tm * 16, k * 32),
                        bfrag_wt(wt_fc1, D_, k * 32, c * 256 + tn * 16), acc);
      int col = tn * 16 + (lane & 15);
      int rb = tm * 16 + ((lane >> 4) << 3);
      float fb = fc1_b[c * 256 + col];
#pragma unroll
      for (int r = 0; r < 8; ++r)
        hc[(rb + r) * D_ + col] = (bf16)gelu_(acc[r] + fb);
    }
    __syncthreads();
#pragma unroll
    for (int i = 0; i < 8; ++i) {
      int t = wave + i * 8;
      int tm = t & 3, tn = t >> 2;
#pragma unroll
      for (int k = 0; k < 8; ++k)
        oacc[i] = wmma_bf16(afrag_lds(hc, D_, tm * 16, k * 32),
                            bfrag_wt(wt_fc2, M_, c * 256 + k * 32, tn * 16),
                            oacc[i]);
    }
    __syncthreads();
  }

  const float* mrow = mask + p * N_;
#pragma unroll
  for (int i = 0; i < 8; ++i) {
    int t = wave + i * 8;
    int tm = t & 3, tn = t >> 2;
    int col = tn * 16 + (lane & 15);
    int rb = tm * 16 + ((lane >> 4) << 3);
    float fb = fc2_b[col];
#pragma unroll
    for (int r = 0; r < 8; ++r) {
      int row = rb + r;
      x5[wbase + (size_t)row * D_ + col] =
          xs[row * D_ + col] + (oacc[i][r] + fb) * mrow[row];
    }
  }
}

// =====================================================================
// Kernel C: masked mean-pool over N + LN(bn1) -> z (bf16) in workspace
// =====================================================================
__global__ __launch_bounds__(256) void k_pool_ln(
    const float* __restrict__ x5, const float* __restrict__ mask,
    const float* __restrict__ g, const float* __restrict__ b,
    bf16* __restrict__ zb) {
  __shared__ float red[8], red2[8];
  int blk = blockIdx.x;  // bt*P + p
  int p = blk & (P_ - 1);
  int d = threadIdx.x;
  const float* xw = x5 + (size_t)blk * N_ * D_;
  float s = 0.f, ms = 0.f;
  for (int n = 0; n < N_; ++n) {
    float mv = mask[p * N_ + n];
    s += xw[(size_t)n * D_ + d] * mv;
    ms += mv;
  }
  float z = s / fmaxf(ms, 1.0f);
  float v = z, v2 = z * z;
#pragma unroll
  for (int m = 16; m >= 1; m >>= 1) {
    v += __shfl_xor(v, m, 32);
    v2 += __shfl_xor(v2, m, 32);
  }
  int lane = threadIdx.x & 31, wave = threadIdx.x >> 5;
  if (lane == 0) { red[wave] = v; red2[wave] = v2; }
  __syncthreads();
  float ts = 0.f, ts2 = 0.f;
#pragma unroll
  for (int i = 0; i < 8; ++i) { ts += red[i]; ts2 += red2[i]; }
  float mu = ts * (1.f / D_);
  float rs = rsqrtf(ts2 * (1.f / D_) - mu * mu + EPS);
  zb[(size_t)blk * D_ + d] = (bf16)((z - mu) * rs * g[d] + b[d]);
}

// =====================================================================
// Kernel D: batch path — attention over P=32 + MLP, one block per (b,t)
// =====================================================================
__global__ __launch_bounds__(256) void k_batch(
    const bf16* __restrict__ zb, const bf16* __restrict__ wt_qkv,
    const float* __restrict__ qkv_b, const bf16* __restrict__ wt_proj,
    const float* __restrict__ proj_b, const bf16* __restrict__ wt_fc1,
    const float* __restrict__ fc1_b, const bf16* __restrict__ wt_fc2,
    const float* __restrict__ fc2_b, float* __restrict__ zf) {
  extern __shared__ __align__(16) char smem[];
  bf16* zl   = (bf16*)smem;                                // 32x256 (16KB)
  bf16* qkvl = (bf16*)(smem + 16384);                      // 32x768 (48KB)
  bf16* Sa   = (bf16*)(smem + 65536) + wave_() * P_ * P_;  // 2KB/wave (16KB)
  bf16* ol   = (bf16*)(smem + 81920);                      // 32x256 (16KB)
  bf16* ab   = (bf16*)(smem + 98304);                      // 32x256 (16KB)
  bf16* hcc  = (bf16*)(smem + 114688);                     // 32x256 (16KB)
  int bt = blockIdx.x;
  int lane = lane_(), wave = wave_();

  const bf16* zsrc = zb + (size_t)bt * P_ * D_;
  for (int i = threadIdx.x; i < P_ * D_; i += 256) zl[i] = zsrc[i];
  __syncthreads();

  // qkv: (32x256)@(256x768), 2x48 tiles
  for (int t = wave; t < 2 * 48; t += 8) {
    int tm = t & 1, tn = t >> 1;
    f32x8 acc = zero8();
#pragma unroll
    for (int k = 0; k < 8; ++k)
      acc = wmma_bf16(afrag_lds(zl, D_, tm * 16, k * 32),
                      bfrag_wt(wt_qkv, D_, k * 32, tn * 16), acc);
    int col = tn * 16 + (lane & 15);
    int rb = tm * 16 + ((lane >> 4) << 3);
    float qb = qkv_b[col];
#pragma unroll
    for (int r = 0; r < 8; ++r)
      qkvl[(rb + r) * (3 * D_) + col] = (bf16)(acc[r] + qb);
  }
  __syncthreads();

  // attention, wave == head, 32 tokens
  {
    int h = wave;
    f32x8 sacc[2][2];
    {
      bf16x16 afr[2];
#pragma unroll
      for (int tm = 0; tm < 2; ++tm)
        afr[tm] = afrag_lds(qkvl, 3 * D_, tm * 16, h * HD);
#pragma unroll
      for (int tn = 0; tn < 2; ++tn) {
        bf16x16 bfr = bfrag_lds_nk(qkvl, 3 * D_, D_ + h * HD, tn * 16);
#pragma unroll
        for (int tm = 0; tm < 2; ++tm)
          sacc[tm][tn] = wmma_bf16(afr[tm], bfr, zero8());
      }
    }
    const float scale = 0.17677669529663687f;
#pragma unroll
    for (int tm = 0; tm < 2; ++tm) {
#pragma unroll
      for (int r = 0; r < 8; ++r) {
        int row = tm * 16 + r + ((lane >> 4) << 3);
        float v0 = sacc[tm][0][r] * scale;
        float v1 = sacc[tm][1][r] * scale;
        float mx = fmaxf(v0, v1);
#pragma unroll
        for (int m = 8; m >= 1; m >>= 1) mx = fmaxf(mx, __shfl_xor(mx, m, 32));
        v0 = expf(v0 - mx);
        v1 = expf(v1 - mx);
        float sum = v0 + v1;
#pragma unroll
        for (int m = 8; m >= 1; m >>= 1) sum += __shfl_xor(sum, m, 32);
        float inv = 1.f / sum;
        Sa[row * P_ + (lane & 15)] = (bf16)(v0 * inv);
        Sa[row * P_ + 16 + (lane & 15)] = (bf16)(v1 * inv);
      }
    }
    // O = A(32x32) @ V(32x32); V-fragments hoisted (2 unique, reused by 2 tm)
    bf16x16 bv[2];
#pragma unroll
    for (int tn = 0; tn < 2; ++tn)
      bv[tn] = bfrag_lds_kn(qkvl, 3 * D_, 0, 2 * D_ + h * HD + tn * 16);
#pragma unroll
    for (int tm = 0; tm < 2; ++tm) {
      bf16x16 as0 = afrag_lds(Sa, P_, tm * 16, 0);
#pragma unroll
      for (int tn = 0; tn < 2; ++tn) {
        f32x8 acc = wmma_bf16(as0, bv[tn], zero8());
        cstore_lds(ol, D_, tm * 16, h * HD + tn * 16, acc);
      }
    }
  }
  __syncthreads();

  // proj -> ab (bf16)
  for (int t = wave; t < 2 * 16; t += 8) {
    int tm = t & 1, tn = t >> 1;
    f32x8 acc = zero8();
#pragma unroll
    for (int k = 0; k < 8; ++k)
      acc = wmma_bf16(afrag_lds(ol, D_, tm * 16, k * 32),
                      bfrag_wt(wt_proj, D_, k * 32, tn * 16), acc);
    int col = tn * 16 + (lane & 15);
    int rb = tm * 16 + ((lane >> 4) << 3);
    float pb = proj_b[col];
#pragma unroll
    for (int r = 0; r < 8; ++r) ab[(rb + r) * D_ + col] = (bf16)(acc[r] + pb);
  }
  __syncthreads();

  // MLP 256->1024->256 in M-chunks
  f32x8 oacc[4];
#pragma unroll
  for (int i = 0; i < 4; ++i) oacc[i] = zero8();
  for (int c = 0; c < 4; ++c) {
    for (int t = wave; t < 2 * 16; t += 8) {
      int tm = t & 1, tn = t >> 1;
      f32x8 acc = zero8();
#pragma unroll
      for (int k = 0; k < 8; ++k)
        acc = wmma_bf16(afrag_lds(ab, D_, tm * 16, k * 32),
                        bfrag_wt(wt_fc1, D_, k * 32, c * 256 + tn * 16), acc);
      int col = tn * 16 + (lane & 15);
      int rb = tm * 16 + ((lane >> 4) << 3);
      float fb = fc1_b[c * 256 + col];
#pragma unroll
      for (int r = 0; r < 8; ++r)
        hcc[(rb + r) * D_ + col] = (bf16)gelu_(acc[r] + fb);
    }
    __syncthreads();
#pragma unroll
    for (int i = 0; i < 4; ++i) {
      int t = wave + i * 8;
      int tm = t & 1, tn = t >> 1;
#pragma unroll
      for (int k = 0; k < 8; ++k)
        oacc[i] = wmma_bf16(afrag_lds(hcc, D_, tm * 16, k * 32),
                            bfrag_wt(wt_fc2, M_, c * 256 + k * 32, tn * 16),
                            oacc[i]);
    }
    __syncthreads();
  }
#pragma unroll
  for (int i = 0; i < 4; ++i) {
    int t = wave + i * 8;
    int tm = t & 1, tn = t >> 1;
    int col = tn * 16 + (lane & 15);
    int rb = tm * 16 + ((lane >> 4) << 3);
    float fb = fc2_b[col];
#pragma unroll
    for (int r = 0; r < 8; ++r)
      zf[((size_t)bt * P_ + rb + r) * D_ + col] = oacc[i][r] + fb;
  }
}

// =====================================================================
// Kernel E: x5 += z broadcast over N
// =====================================================================
__global__ __launch_bounds__(256) void k_bcast_add(float* __restrict__ out,
                                                   const float* __restrict__ zf) {
  size_t idx = (size_t)blockIdx.x * blockDim.x + threadIdx.x;
  int d = (int)(idx & (size_t)(D_ - 1));
  size_t btp = (idx >> 8) >> 6;  // /D /N
  out[idx] += zf[btp * D_ + d];
}

extern "C" void kernel_launch(void* const* d_in, const int* in_sizes, int n_in,
                              void* d_out, int out_size, void* d_ws,
                              size_t ws_size, hipStream_t stream) {
  (void)in_sizes; (void)n_in; (void)out_size; (void)ws_size;
  const float* x        = (const float*)d_in[0];
  const float* mask     = (const float*)d_in[1];
  const float* adj      = (const float*)d_in[2];
  const float* dn1_g    = (const float*)d_in[3];
  const float* dn1_b    = (const float*)d_in[4];
  const float* dn2_g    = (const float*)d_in[5];
  const float* dn2_b    = (const float*)d_in[6];
  const float* bn1_g    = (const float*)d_in[7];
  const float* bn1_b    = (const float*)d_in[8];
  const float* d_qkv_w  = (const float*)d_in[9];
  const float* d_qkv_b  = (const float*)d_in[10];
  const float* d_proj_w = (const float*)d_in[11];
  const float* d_proj_b = (const float*)d_in[12];
  const float* d_fc1_w  = (const float*)d_in[13];
  const float* d_fc1_b  = (const float*)d_in[14];
  const float* d_fc2_w  = (const float*)d_in[15];
  const float* d_fc2_b  = (const float*)d_in[16];
  const float* b_qkv_w  = (const float*)d_in[17];
  const float* b_qkv_b  = (const float*)d_in[18];
  const float* b_proj_w = (const float*)d_in[19];
  const float* b_proj_b = (const float*)d_in[20];
  const float* b_fc1_w  = (const float*)d_in[21];
  const float* b_fc1_b  = (const float*)d_in[22];
  const float* b_fc2_w  = (const float*)d_in[23];
  const float* b_fc2_b  = (const float*)d_in[24];
  float* out = (float*)d_out;

  // workspace layout: 8 transposed bf16 weights, then zb (bf16), zf (f32)
  char* ws = (char*)d_ws;
  const size_t SZ_QKV = (size_t)D_ * 3 * D_ * 2;   // 393216
  const size_t SZ_PROJ = (size_t)D_ * D_ * 2;      // 131072
  const size_t SZ_FC = (size_t)D_ * M_ * 2;        // 524288
  bf16* wt_qkv_d  = (bf16*)(ws);
  bf16* wt_proj_d = (bf16*)(ws + SZ_QKV);
  bf16* wt_fc1_d  = (bf16*)(ws + SZ_QKV + SZ_PROJ);
  bf16* wt_fc2_d  = (bf16*)(ws + SZ_QKV + SZ_PROJ + SZ_FC);
  bf16* wt_qkv_b  = (bf16*)(ws + SZ_QKV + SZ_PROJ + 2 * SZ_FC);
  bf16* wt_proj_b = (bf16*)(ws + 2 * SZ_QKV + SZ_PROJ + 2 * SZ_FC);
  bf16* wt_fc1_b  = (bf16*)(ws + 2 * SZ_QKV + 2 * SZ_PROJ + 2 * SZ_FC);
  bf16* wt_fc2_b  = (bf16*)(ws + 2 * SZ_QKV + 2 * SZ_PROJ + 3 * SZ_FC);
  char* tail      = ws + 2 * SZ_QKV + 2 * SZ_PROJ + 4 * SZ_FC;
  bf16* zbuf = (bf16*)tail;                               // 1024*256 bf16
  float* zf = (float*)(tail + (size_t)1024 * 256 * 2);    // 1024*256 f32

  // weight prep (runs every call; deterministic)
  k_wt<<<768, 256, 0, stream>>>(d_qkv_w, wt_qkv_d, D_, 3 * D_);
  k_wt<<<256, 256, 0, stream>>>(d_proj_w, wt_proj_d, D_, D_);
  k_wt<<<1024, 256, 0, stream>>>(d_fc1_w, wt_fc1_d, D_, M_);
  k_wt<<<1024, 256, 0, stream>>>(d_fc2_w, wt_fc2_d, M_, D_);
  k_wt<<<768, 256, 0, stream>>>(b_qkv_w, wt_qkv_b, D_, 3 * D_);
  k_wt<<<256, 256, 0, stream>>>(b_proj_w, wt_proj_b, D_, D_);
  k_wt<<<1024, 256, 0, stream>>>(b_fc1_w, wt_fc1_b, D_, M_);
  k_wt<<<1024, 256, 0, stream>>>(b_fc2_w, wt_fc2_b, M_, D_);

  const int nwin = B_ * T_ * P_;  // 1024 windows
  k_win_attn<<<nwin, 256, 262144, stream>>>(x, mask, adj, dn1_g, dn1_b,
                                            wt_qkv_d, d_qkv_b, wt_proj_d,
                                            d_proj_b, out);
  k_win_mlp<<<nwin, 256, 131072, stream>>>(out, mask, dn2_g, dn2_b, wt_fc1_d,
                                           d_fc1_b, wt_fc2_d, d_fc2_b);
  k_pool_ln<<<nwin, 256, 0, stream>>>(out, mask, bn1_g, bn1_b, zbuf);
  k_batch<<<B_ * T_, 256, 131072, stream>>>(zbuf, wt_qkv_b, b_qkv_b, wt_proj_b,
                                            b_proj_b, wt_fc1_b, b_fc1_b,
                                            wt_fc2_b, b_fc2_b, zf);
  const int total = B_ * T_ * P_ * N_ * D_;  // 16,777,216
  k_bcast_add<<<total / 256, 256, 0, stream>>>(out, zf);
}
